// SingleHeadWaveletAttention_30700426231867
// MI455X (gfx1250) — matrix operand order repaired and checked by reading, not compiled, not run on hardware
//
#include <hip/hip_runtime.h>

// CDNA5 / gfx1250 wave32 WMMA implementation of SingleHeadWaveletAttention.
// All heavy matmuls use v_wmma_f32_16x16x32_bf16 via one generic tiled GEMM
// with vectorized (b128) LDS fragment loads.

typedef __attribute__((ext_vector_type(16))) __bf16 v16bf;
typedef __attribute__((ext_vector_type(8)))  float  v8f;

namespace {
constexpr int Bn = 4;
constexpr int Tn = 2048;
constexpr int Cn = 768;
constexpr float SCALE = 0.036084391824351615f; // 1/sqrt(768)
}

// ---------------------------------------------------------------------------
// Generic WMMA GEMM:  C(f32, MxN) = A(bf16, MxK row-major) * op(B)
//   TRANSB=true : B stored N x K row-major (computes A @ B^T)
//   TRANSB=false: B stored K x N row-major (computes A @ B)
//   BANDSKIP    : skip tiles fully outside {0 <= m-n <= 64} band
// Block tile 64x128, 8 waves (4 M-subtiles x 2 N-subtiles of 16x64), K-step 32.
// A is staged into LDS in contiguous k order (k = 16*lhalf + s) so each
// lane's 16-element WMMA fragment is one contiguous 32B read (2x ds_load_b128).
// B is staged n-major so B fragments are likewise contiguous 32B reads.
// ---------------------------------------------------------------------------
template <bool TRANSB, bool BANDSKIP>
__global__ __launch_bounds__(256) void gemm_wmma(
    const __bf16* __restrict__ A, const __bf16* __restrict__ Bm,
    float* __restrict__ C, int M, int N, int K,
    long long As, long long Bs, long long Cs) {
  const int bz = blockIdx.z;
  A += (long long)bz * As;
  Bm += (long long)bz * Bs;
  C += (long long)bz * Cs;
  const int m0 = blockIdx.y * 64;
  const int n0 = blockIdx.x * 128;
  if (BANDSKIP) {
    if (n0 > m0 + 63) return;   // whole tile above the causal diagonal
    if (m0 >= n0 + 192) return; // whole tile strictly beyond the near band
  }
  __shared__ alignas(32) __bf16 Abuf[64][48];  // [m][k], 96B stride
  __shared__ alignas(32) __bf16 Bbuf[128][48]; // [n][k], 96B stride

  const int tid = threadIdx.x;
  const int lane = tid & 31;
  const int wave = tid >> 5;
  const int wm = (wave >> 1) * 16; // 0,16,32,48
  const int wn = (wave & 1) * 64;  // 0,64
  const int l16 = lane & 15;
  const int lhalf = lane >> 4;

  v8f acc[4] = {{}, {}, {}, {}};

  for (int k0 = 0; k0 < K; k0 += 32) {
    { // stage A tile 64x32: one 16B load + one 16B store per thread
      const int r = tid >> 2;
      const int cb = (tid & 3) * 8; // 0,8,16,24
      const int dcb = cb; // contiguous k (matches B fragment order)
      const uint4* src = (const uint4*)(A + (long long)(m0 + r) * K + k0 + cb);
      if (k0 + 32 < K) __builtin_prefetch(src + 2, 0, 1);
      *(uint4*)&Abuf[r][dcb] = *src;
    }
    if (TRANSB) { // B is NxK: 128 n-rows x 32 k, 16 elems per thread
      const int r = tid >> 1;        // local n 0..127
      const int cb = (tid & 1) * 16; // local k 0 or 16
      const uint4* src = (const uint4*)(Bm + (long long)(n0 + r) * K + k0 + cb);
      if (k0 + 32 < K) __builtin_prefetch(src + 2, 0, 1);
      *(uint4*)&Bbuf[r][cb] = src[0];
      *(uint4*)&Bbuf[r][cb + 8] = src[1];
    } else { // B is KxN: wide reads along n, transpose-scatter into [n][k]
      const int r = tid >> 3;         // local k 0..31
      const int cb = (tid & 7) * 16;  // local n 0..112
      const uint4* src = (const uint4*)(Bm + (long long)(k0 + r) * N + n0 + cb);
      if (k0 + 32 < K) __builtin_prefetch((const char*)src + (long long)64 * N, 0, 1);
      uint4 d0 = src[0], d1 = src[1];
      const __bf16* e0 = (const __bf16*)&d0;
      const __bf16* e1 = (const __bf16*)&d1;
#pragma unroll
      for (int i = 0; i < 8; ++i) Bbuf[cb + i][r] = e0[i];
#pragma unroll
      for (int i = 0; i < 8; ++i) Bbuf[cb + 8 + i][r] = e1[i];
    }
    __syncthreads();

    // A fragment: one contiguous 32B read thanks to contiguous k staging.
    const v16bf af = *(const v16bf*)&Abuf[wm + l16][lhalf * 16];
#pragma unroll
    for (int j = 0; j < 4; ++j) {
      const v16bf bfr = *(const v16bf*)&Bbuf[wn + j * 16 + l16][lhalf * 16];
      acc[j] = __builtin_amdgcn_wmma_f32_16x16x32_bf16(
          false, af, false, bfr, (short)0, acc[j], false, false);
    }
    __syncthreads();
  }

  // C/D layout: lane l16 -> N=l16; VGPR r -> M = r + 8*lhalf
#pragma unroll
  for (int j = 0; j < 4; ++j) {
    const int n = n0 + wn + j * 16 + l16;
#pragma unroll
    for (int r = 0; r < 8; ++r) {
      const int m = m0 + wm + r + lhalf * 8;
      C[(long long)m * N + n] = acc[j][r];
    }
  }
}

// ---------------------------------------------------------------------------
// Gauss-Jordan inverse of W_Q (768x768), writing inv(W_Q) directly as bf16
// K x N operand for the k-projection (k = x @ inv(W_Q)). Single block.
// ---------------------------------------------------------------------------
__global__ __launch_bounds__(1024) void invert_kernel(
    const float* __restrict__ Wq, float* __restrict__ aug,
    __bf16* __restrict__ Bk) {
  constexpr int N = 768, W = 1536;
  const int tid = threadIdx.x;
  __shared__ float fac[N];
  for (long long i = tid; i < (long long)N * W; i += 1024) {
    int r = (int)(i / W), c = (int)(i % W);
    aug[i] = (c < N) ? Wq[(long long)r * N + c] : ((c - N) == r ? 1.f : 0.f);
  }
  __syncthreads();
  for (int p = 0; p < N; ++p) {
    float piv = aug[(long long)p * W + p];
    __syncthreads();
    float ip = 1.0f / piv;
    for (int c = tid; c < W; c += 1024) aug[(long long)p * W + c] *= ip;
    for (int r = tid; r < N; r += 1024)
      if (r != p) fac[r] = aug[(long long)r * W + p];
    __syncthreads();
    for (long long i = tid; i < (long long)N * W; i += 1024) {
      int r = (int)(i / W), c = (int)(i % W);
      if (r != p) aug[i] -= fac[r] * aug[(long long)p * W + c];
    }
    __syncthreads();
  }
  for (long long i = tid; i < (long long)N * N; i += 1024) {
    int r = (int)(i / N), c = (int)(i % N);
    Bk[i] = (__bf16)aug[(long long)r * W + N + c];
  }
}

__global__ __launch_bounds__(256) void f32_to_bf16(
    const float* __restrict__ in, __bf16* __restrict__ out, long long n) {
  long long i = (long long)blockIdx.x * blockDim.x + threadIdx.x;
  const long long stride = (long long)gridDim.x * blockDim.x;
  for (; i < n; i += stride) out[i] = (__bf16)in[i];
}

// ---------------------------------------------------------------------------
// RoPE + L2 normalize, in-place f32 + bf16 copy. One block per (b,t) row.
// ---------------------------------------------------------------------------
__global__ __launch_bounds__(256) void rope_l2norm(
    float* __restrict__ data, __bf16* __restrict__ outb) {
  const long long row = blockIdx.x; // b*Tn + t
  const int t = (int)(row % Tn);
  float* src = data + row * Cn;
  __shared__ float buf[Cn];
  __shared__ float red[8];
  const int tid = threadIdx.x;
  constexpr float LOG10K = 9.210340371976184f; // ln(10000)
  for (int i = tid; i < 384; i += 256) {
    float invf = __expf(-LOG10K * ((float)i / 384.0f));
    float c, s;
    __sincosf((float)t * invf, &s, &c); // s=sin, c=cos
    float t1 = src[i], t2 = src[i + 384];
    buf[i] = t1 * c - t2 * s;
    buf[i + 384] = t1 * s + t2 * c;
  }
  __syncthreads();
  float ss = 0.f;
  for (int i = tid; i < Cn; i += 256) { float x = buf[i]; ss += x * x; }
  for (int o = 16; o > 0; o >>= 1) ss += __shfl_down(ss, o, 32);
  if ((tid & 31) == 0) red[tid >> 5] = ss;
  __syncthreads();
  if (tid == 0) {
    float tot = 0.f;
    for (int w = 0; w < 8; ++w) tot += red[w];
    red[0] = 1.0f / (sqrtf(tot) + 1e-8f);
  }
  __syncthreads();
  const float inv = red[0];
  for (int i = tid; i < Cn; i += 256) {
    float v = buf[i] * inv;
    src[i] = v;
    outb[row * Cn + i] = (__bf16)v;
  }
}

// ---------------------------------------------------------------------------
// qf[b][m][c] = sum_t q[b][t][c] * Wh[t][m]  (Haar columns analytic)
// ---------------------------------------------------------------------------
__global__ __launch_bounds__(256) void haar_project(
    const float* __restrict__ qn, float* __restrict__ qf) {
  const int m = blockIdx.x, b = blockIdx.y;
  int level, kblk;
  if (m == 0) { level = 0; kblk = 0; }
  else if (m < 3) { level = 1; kblk = m - 1; }
  else { level = 2; kblk = m - 3; }
  const int bs = Tn >> level;
  const int half = bs >> 1;
  const float val = rsqrtf((float)half);
  const int s0 = kblk * bs;
  const float* base = qn + (long long)b * Tn * Cn;
  for (int c = threadIdx.x; c < Cn; c += 256) {
    float pos = 0.f, neg = 0.f;
    for (int t = 0; t < half; ++t) pos += base[(long long)(s0 + t) * Cn + c];
    for (int t = 0; t < half; ++t) neg += base[(long long)(s0 + half + t) * Cn + c];
    qf[((long long)b * 7 + m) * Cn + c] = (pos - neg) * val;
  }
}

// comp[b][m][n] = SCALE * dot(qf[b][m], kf[b][n])
__global__ __launch_bounds__(256) void comp_kernel(
    const float* __restrict__ qf, const float* __restrict__ kf,
    float* __restrict__ comp) {
  const int mn = blockIdx.x, b = blockIdx.y;
  const int m = mn / 7, n = mn % 7;
  const float* a = qf + ((long long)b * 7 + m) * Cn;
  const float* c2 = kf + ((long long)b * 7 + n) * Cn;
  float ss = 0.f;
  for (int i = threadIdx.x; i < Cn; i += 256) ss += a[i] * c2[i];
  __shared__ float red[8];
  for (int o = 16; o > 0; o >>= 1) ss += __shfl_down(ss, o, 32);
  if ((threadIdx.x & 31) == 0) red[threadIdx.x >> 5] = ss;
  __syncthreads();
  if (threadIdx.x == 0) {
    float tot = 0.f;
    for (int w = 0; w < 8; ++w) tot += red[w];
    comp[((long long)b * 7 + m) * 7 + n] = tot * SCALE;
  }
}

// Haar row t: 3 nonzero (col, value) pairs.
__device__ __forceinline__ void haar_row(int t, int* cols, float* vals) {
  cols[0] = 0;
  vals[0] = (t < 1024 ? 1.f : -1.f) * 0.03125f;                 // 1/sqrt(1024)
  cols[1] = 1 + (t >> 10);
  vals[1] = ((t & 1023) < 512 ? 1.f : -1.f) * 0.044194173824159216f; // 1/sqrt(512)
  cols[2] = 3 + (t >> 9);
  vals[2] = ((t & 511) < 256 ? 1.f : -1.f) * 0.0625f;           // 1/sqrt(256)
}

// S[b][t][s] (s<=t): near -> *=SCALE, far -> Wh[t] @ comp @ Wh[s]
__global__ __launch_bounds__(256) void mix_kernel(
    float* __restrict__ S, const float* __restrict__ comp) {
  const int t = blockIdx.x, b = blockIdx.y;
  __shared__ float u[7];
  const float* cp = comp + (long long)b * 49;
  if (threadIdx.x < 7) {
    int tc[3]; float tv[3];
    haar_row(t, tc, tv);
    float acc = 0.f;
    for (int j = 0; j < 3; ++j) acc += tv[j] * cp[tc[j] * 7 + threadIdx.x];
    u[threadIdx.x] = acc;
  }
  __syncthreads();
  float* row = S + ((long long)b * Tn + t) * (long long)Tn;
  for (int s = threadIdx.x; s <= t; s += 256) {
    if (t - s <= 64) {
      row[s] *= SCALE;
    } else {
      int sc[3]; float sv[3];
      haar_row(s, sc, sv);
      row[s] = u[sc[0]] * sv[0] + u[sc[1]] * sv[1] + u[sc[2]] * sv[2];
    }
  }
}

__device__ __forceinline__ float blk_sum(float v, float* red) {
  for (int o = 16; o > 0; o >>= 1) v += __shfl_down(v, o, 32);
  const int lane = threadIdx.x & 31, w = threadIdx.x >> 5;
  if (lane == 0) red[w] = v;
  __syncthreads();
  float r = (threadIdx.x < 8) ? red[threadIdx.x] : 0.f;
  r += __shfl_down(r, 4, 32);
  r += __shfl_down(r, 2, 32);
  r += __shfl_down(r, 1, 32);
  if (threadIdx.x == 0) red[0] = r;
  __syncthreads();
  float out = red[0];
  __syncthreads();
  return out;
}

__device__ __forceinline__ float blk_max(float v, float* red) {
  for (int o = 16; o > 0; o >>= 1) v = fmaxf(v, __shfl_down(v, o, 32));
  const int lane = threadIdx.x & 31, w = threadIdx.x >> 5;
  if (lane == 0) red[w] = v;
  __syncthreads();
  float r = (threadIdx.x < 8) ? red[threadIdx.x] : -3.4e38f;
  r = fmaxf(r, __shfl_down(r, 4, 32));
  r = fmaxf(r, __shfl_down(r, 2, 32));
  r = fmaxf(r, __shfl_down(r, 1, 32));
  if (threadIdx.x == 0) red[0] = r;
  __syncthreads();
  float out = red[0];
  __syncthreads();
  return out;
}

// Variance-scaled softmax over s in [0,t]; writes bf16 probabilities (0 beyond t)
__global__ __launch_bounds__(256) void vss_softmax(
    const float* __restrict__ S, __bf16* __restrict__ P) {
  const int t = blockIdx.x, b = blockIdx.y;
  const float* row = S + ((long long)b * Tn + t) * (long long)Tn;
  __bf16* prow = P + ((long long)b * Tn + t) * (long long)Tn;
  __shared__ float buf[Tn];
  __shared__ float red[8];
  const int tid = threadIdx.x;
  const int n = t + 1;
  for (int s = tid; s < n; s += 256) buf[s] = row[s];
  __syncthreads();
  float sum = 0.f, sq = 0.f;
  for (int s = tid; s < n; s += 256) { float x = buf[s]; sum += x; sq += x * x; }
  const float tot = blk_sum(sum, red);
  const float tot2 = blk_sum(sq, red);
  const float mean = tot / (float)n;
  const float var = tot2 / (float)n - mean * mean;
  const float istd = 1.0f / sqrtf(fmaxf(var, 1e-6f));
  float mx = -3.4e38f;
  for (int s = tid; s < n; s += 256) mx = fmaxf(mx, (buf[s] - mean) * istd);
  const float gmx = blk_max(mx, red);
  float es = 0.f;
  for (int s = tid; s < n; s += 256) {
    float e = __expf((buf[s] - mean) * istd - gmx);
    buf[s] = e;
    es += e;
  }
  const float z = blk_sum(es, red);
  const float iz = 1.0f / z;
  for (int s = tid; s < n; s += 256) prow[s] = (__bf16)(buf[s] * iz);
  for (int s = n + tid; s < Tn; s += 256) prow[s] = (__bf16)0.0f;
}

// ---------------------------------------------------------------------------
extern "C" void kernel_launch(void* const* d_in, const int* in_sizes, int n_in,
                              void* d_out, int out_size, void* d_ws,
                              size_t ws_size, hipStream_t stream) {
  (void)in_sizes; (void)n_in; (void)out_size; (void)ws_size;
  const float* x = (const float*)d_in[0];
  const float* WQ = (const float*)d_in[1];
  const float* WV = (const float*)d_in[2];
  const float* WO = (const float*)d_in[3];
  float* out = (float*)d_out;

  char* w = (char*)d_ws;
  size_t off = 0;
  auto take = [&](size_t bytes) -> char* {
    char* p = w + off;
    off = (off + bytes + 255) & ~(size_t)255;
    return p;
  };
  const size_t BTC = (size_t)Bn * Tn * Cn;
  const size_t BTT = (size_t)Bn * Tn * Tn;
  float* aug = (float*)take((size_t)768 * 1536 * 4);
  float* q = (float*)take(BTC * 4);
  float* k = (float*)take(BTC * 4);
  float* v = (float*)take(BTC * 4);
  float* S = (float*)take(BTT * 4);
  float* qf = (float*)take((size_t)Bn * 7 * Cn * 4);
  float* kf = (float*)take((size_t)Bn * 7 * Cn * 4);
  float* comp = (float*)take((size_t)Bn * 49 * 4);
  __bf16* xb = (__bf16*)take(BTC * 2);
  __bf16* qb = (__bf16*)take(BTC * 2);
  __bf16* kb = (__bf16*)take(BTC * 2);
  __bf16* vb = (__bf16*)take(BTC * 2);
  __bf16* Bq = (__bf16*)take((size_t)Cn * Cn * 2);
  __bf16* Bk = (__bf16*)take((size_t)Cn * Cn * 2);
  __bf16* Bv = (__bf16*)take((size_t)Cn * Cn * 2);
  __bf16* Bo = (__bf16*)take((size_t)Cn * Cn * 2);
  __bf16* P = (__bf16*)take(BTT * 2);
  float* y = q;     // q (f32) is dead after haar/comp -> reuse for y
  __bf16* yb = xb;  // x_bf16 is dead after projections -> reuse for y_bf16

  // 1) inv(W_Q) -> Bk (bf16, KxN)
  invert_kernel<<<1, 1024, 0, stream>>>(WQ, aug, Bk);
  // 2) bf16 conversions of x and NT weight operands (already NxK row-major)
  f32_to_bf16<<<2048, 256, 0, stream>>>(x, xb, (long long)BTC);
  f32_to_bf16<<<512, 256, 0, stream>>>(WQ, Bq, (long long)Cn * Cn);
  f32_to_bf16<<<512, 256, 0, stream>>>(WV, Bv, (long long)Cn * Cn);
  f32_to_bf16<<<512, 256, 0, stream>>>(WO, Bo, (long long)Cn * Cn);
  // 3) projections
  dim3 gp(Cn / 128, (Bn * Tn) / 64, 1);
  gemm_wmma<true, false><<<gp, 256, 0, stream>>>(xb, Bq, q, Bn * Tn, Cn, Cn, 0, 0, 0);
  gemm_wmma<false, false><<<gp, 256, 0, stream>>>(xb, Bk, k, Bn * Tn, Cn, Cn, 0, 0, 0);
  gemm_wmma<true, false><<<gp, 256, 0, stream>>>(xb, Bv, v, Bn * Tn, Cn, Cn, 0, 0, 0);
  f32_to_bf16<<<2048, 256, 0, stream>>>(v, vb, (long long)BTC);
  // 4) RoPE + L2 norm (in-place f32, bf16 copies)
  rope_l2norm<<<Bn * Tn, 256, 0, stream>>>(q, qb);
  rope_l2norm<<<Bn * Tn, 256, 0, stream>>>(k, kb);
  // 5) Haar compression + comp
  haar_project<<<dim3(7, Bn), 256, 0, stream>>>(q, qf);
  haar_project<<<dim3(7, Bn), 256, 0, stream>>>(k, kf);
  comp_kernel<<<dim3(49, Bn), 256, 0, stream>>>(qf, kf, comp);
  // 6) near scores: banded q @ k^T (band-skipped GEMM)
  dim3 gs(Tn / 128, Tn / 64, Bn);
  gemm_wmma<true, true><<<gs, 256, 0, stream>>>(
      qb, kb, S, Tn, Tn, Cn, (long long)Tn * Cn, (long long)Tn * Cn,
      (long long)Tn * Tn);
  // 7) mix near/far + scale, then variance-scaled softmax -> P (bf16)
  mix_kernel<<<dim3(Tn, Bn), 256, 0, stream>>>(S, comp);
  vss_softmax<<<dim3(Tn, Bn), 256, 0, stream>>>(S, P);
  // 8) y = P @ v
  dim3 gv(Cn / 128, Tn / 64, Bn);
  gemm_wmma<false, false><<<gv, 256, 0, stream>>>(
      P, vb, y, Tn, Cn, Tn, (long long)Tn * Tn, (long long)Tn * Cn,
      (long long)Tn * Cn);
  f32_to_bf16<<<2048, 256, 0, stream>>>(y, yb, (long long)BTC);
  // 9) out = y @ W_O^T
  gemm_wmma<true, false><<<gp, 256, 0, stream>>>(yb, Bo, out, Bn * Tn, Cn, Cn, 0, 0, 0);
}
